// AE_90950227460250
// MI455X (gfx1250) — matrix operand order repaired and checked by reading, not compiled, not run on hardware
//
#include <hip/hip_runtime.h>

// ---------------------------------------------------------------------------
// GCN autoencoder for gfx1250 (MI455X).
//
// Per layer:   h = act @ W            -> WMMA bf16 GEMM (fp32 accumulate)
//              agg = b + self_norm*h  -> streaming init pass
//              agg += scatter(norm*h) -> edge-parallel waves, HW f32 atomics
//              act' = relu(agg)       -> fused relu + fp32->bf16 convert
//
// MI455X roofline: aggregation traffic (~14 GB over 8 layers) dominates at
// 23.3 TB/s (~0.6 ms). GEMMs (~26 G MACs) run on v_wmma_f32_16x16x32_bf16.
// Activations are kept as bf16 in the exact A-fragment layout (padded row
// stride, zero tail) so the GEMM inner loop is pure b128 loads + WMMA:
// no conversion or clamping VALU in the hot path.
// ---------------------------------------------------------------------------

#define NN  50000
#define EE  800000
#define FF  500
#define HH  256
#define CC  10

typedef __bf16          v16bf __attribute__((ext_vector_type(16)));
typedef float           v8f   __attribute__((ext_vector_type(8)));
typedef unsigned short  us16  __attribute__((ext_vector_type(16)));
typedef unsigned short  us8   __attribute__((ext_vector_type(8)));

struct bfrag { us8 lo, hi; };   // 32 bytes == v16bf

__device__ __forceinline__ unsigned short f2bf(float f) {
    unsigned u = __builtin_bit_cast(unsigned, f);
    u += 0x7FFFu + ((u >> 16) & 1u);          // round-to-nearest-even
    return (unsigned short)(u >> 16);
}

// ---------------------------------------------------------------------------
// Degree / normalization
// ---------------------------------------------------------------------------
__global__ void fill_ones_kernel(float* p, int n) {
    int i = blockIdx.x * blockDim.x + threadIdx.x;
    if (i < n) p[i] = 1.0f;                    // deg starts at 1 (self loop)
}

__global__ void deg_acc_kernel(float* deg, const int* __restrict__ dst, int e) {
    int i = blockIdx.x * blockDim.x + threadIdx.x;
    if (i < e)
        __hip_atomic_fetch_add(&deg[dst[i]], 1.0f,
                               __ATOMIC_RELAXED, __HIP_MEMORY_SCOPE_AGENT);
}

__global__ void deg_fin_kernel(float* dinv, float* self_norm, int n) {
    int i = blockIdx.x * blockDim.x + threadIdx.x;
    if (i < n) {
        float r = rsqrtf(dinv[i]);
        dinv[i] = r;
        self_norm[i] = r * r;
    }
}

// ---------------------------------------------------------------------------
// Weight swizzle: row-major fp32 W[din][dout] -> bf16 B-fragment layout.
//   lane n in [0,16):  b[j] = W[kt*32 + j + (j>=8 ? 8 : 0)    ][nt*16 + n]
//   lane n in [16,32): b[j] = W[kt*32 + 8 + j + (j>=8 ? 8 : 0)][nt*16 + n-16]
// Stored as [kt][nt][lane][16]: each lane reads 32 contiguous bytes.
// ---------------------------------------------------------------------------
__global__ void swizzle_w_kernel(const float* __restrict__ W,
                                 unsigned short* __restrict__ out,
                                 int din, int dout, int Kpad, int Npad) {
    int idx = blockIdx.x * blockDim.x + threadIdx.x;
    int total = (Kpad >> 5) * (Npad >> 4) * 512;
    if (idx >= total) return;
    int j     = idx & 15;
    int lane  = (idx >> 4) & 31;
    int t     = idx >> 9;
    int ntils = Npad >> 4;
    int nt    = t % ntils;
    int kt    = t / ntils;
    int k     = kt * 32 + ((lane >> 4) ? 8 : 0) + j + ((j >= 8) ? 8 : 0);
    int n     = nt * 16 + (lane & 15);
    float v = (k < din && n < dout) ? W[(size_t)k * dout + n] : 0.0f;
    out[idx] = f2bf(v);
}

// ---------------------------------------------------------------------------
// Activation pad+convert: fp32 [N x din] -> bf16 [N x ldk], zero tail.
// grid = (ceil(ldk/256), N)
// ---------------------------------------------------------------------------
__global__ void pad_cvt_kernel(unsigned short* __restrict__ dst,
                               const float* __restrict__ src,
                               int din, int ldk) {
    int j = blockIdx.x * blockDim.x + threadIdx.x;
    int i = blockIdx.y;
    if (j >= ldk) return;
    float v = (j < din) ? src[(size_t)i * din + j] : 0.0f;
    dst[(size_t)i * ldk + j] = f2bf(v);
}

// ---------------------------------------------------------------------------
// GEMM: out[M x dout] = act_bf16[M x ldk] @ W  via v_wmma_f32_16x16x32_bf16.
// Each wave computes two adjacent 16x16 tiles sharing one A fragment.
// A fragment = two aligned b128 loads; B fragments = pre-swizzled b128 pairs.
// ---------------------------------------------------------------------------
__global__ __launch_bounds__(256)
void gemm_wmma_kernel(const unsigned short* __restrict__ act,
                      const unsigned short* __restrict__ wswz,
                      float* __restrict__ out,
                      int ldk, int dout, int ktiles, int Npad) {
    const int lane   = threadIdx.x & 31;
    const int wave   = threadIdx.x >> 5;
    const int mtile  = blockIdx.x;
    const int ntile0 = (blockIdx.y * 8 + wave) * 2;
    if (ntile0 * 16 >= Npad) return;               // wave-uniform, EXEC all-1
    const bool has2  = ((ntile0 + 1) * 16 < Npad); // wave-uniform
    const int ntils  = Npad >> 4;

    const unsigned short* abase =
        act + (size_t)(mtile * 16 + (lane & 15)) * ldk + ((lane >> 4) ? 8 : 0);
    const unsigned short* bbase = wswz + ((size_t)ntile0 * 32 + lane) * 16;
    const size_t bstep = (size_t)ntils * 512;      // ushorts per kt

    v8f c0 = {}, c1 = {};
    if (has2) {
#pragma unroll 2
        for (int kt = 0; kt < ktiles; ++kt) {
            bfrag af;
            af.lo = *(const us8*)(abase + kt * 32);
            af.hi = *(const us8*)(abase + kt * 32 + 16);
            v16bf a  = __builtin_bit_cast(v16bf, af);
            v16bf b0 = __builtin_bit_cast(v16bf, *(const us16*)(bbase + kt * bstep));
            v16bf b1 = __builtin_bit_cast(v16bf, *(const us16*)(bbase + kt * bstep + 512));
            c0 = __builtin_amdgcn_wmma_f32_16x16x32_bf16(
                     false, a, false, b0, (short)0, c0, false, false);
            c1 = __builtin_amdgcn_wmma_f32_16x16x32_bf16(
                     false, a, false, b1, (short)0, c1, false, false);
        }
    } else {
#pragma unroll 2
        for (int kt = 0; kt < ktiles; ++kt) {
            bfrag af;
            af.lo = *(const us8*)(abase + kt * 32);
            af.hi = *(const us8*)(abase + kt * 32 + 16);
            v16bf a  = __builtin_bit_cast(v16bf, af);
            v16bf b0 = __builtin_bit_cast(v16bf, *(const us16*)(bbase + kt * bstep));
            c0 = __builtin_amdgcn_wmma_f32_16x16x32_bf16(
                     false, a, false, b0, (short)0, c0, false, false);
        }
    }

    const int mbase = mtile * 16 + ((lane >> 4) ? 8 : 0);
    const int n0 = ntile0 * 16 + (lane & 15);
    if (n0 < dout) {
#pragma unroll
        for (int r = 0; r < 8; ++r)
            out[(size_t)(mbase + r) * dout + n0] = c0[r];
    }
    if (has2) {
        const int n1 = n0 + 16;
        if (n1 < dout) {
#pragma unroll
            for (int r = 0; r < 8; ++r)
                out[(size_t)(mbase + r) * dout + n1] = c1[r];
        }
    }
}

// ---------------------------------------------------------------------------
// Aggregation
// ---------------------------------------------------------------------------
// grid = (ceil(dout/256), N)
__global__ void agg_init_kernel(float* __restrict__ out,
                                const float* __restrict__ h,
                                const float* __restrict__ bias,
                                const float* __restrict__ self_norm,
                                int dout) {
    int j = blockIdx.x * blockDim.x + threadIdx.x;
    int i = blockIdx.y;
    if (j >= dout) return;
    size_t idx = (size_t)i * dout + j;
    out[idx] = bias[j] + self_norm[i] * h[idx];
}

__global__ void scatter_kernel(float* __restrict__ out,
                               const float* __restrict__ h,
                               const int* __restrict__ src,
                               const int* __restrict__ dst,
                               const float* __restrict__ dinv,
                               int e, int dout) {
    int gid  = blockIdx.x * blockDim.x + threadIdx.x;
    int edge = gid >> 5;
    int lane = gid & 31;
    if (edge >= e) return;
    int s = src[edge], d = dst[edge];
    float w = dinv[s] * dinv[d];
    const float* hs = h   + (size_t)s * dout;
    float*       od = out + (size_t)d * dout;
    for (int j = lane; j < dout; j += 32)
        __hip_atomic_fetch_add(&od[j], hs[j] * w,
                               __ATOMIC_RELAXED, __HIP_MEMORY_SCOPE_AGENT);
}

// relu + bf16 convert; optionally also writes fp32 copy (for d_out slices).
// Requires dout == ldk (true for all relu layers: H=256).
__global__ void relu_cvt_kernel(unsigned short* __restrict__ dstbf,
                                float* __restrict__ dst32, int write32,
                                const float* __restrict__ src, long long n) {
    long long i = (long long)blockIdx.x * blockDim.x + threadIdx.x;
    if (i >= n) return;
    float r = fmaxf(src[i], 0.0f);
    dstbf[i] = f2bf(r);
    if (write32) dst32[i] = r;
}

// ---------------------------------------------------------------------------
// Host launcher
// ---------------------------------------------------------------------------
extern "C" void kernel_launch(void* const* d_in, const int* in_sizes, int n_in,
                              void* d_out, int out_size, void* d_ws, size_t ws_size,
                              hipStream_t stream) {
    (void)in_sizes; (void)n_in; (void)out_size; (void)ws_size;

    const float* x   = (const float*)d_in[0];
    const int*   src = (const int*)d_in[1];
    const int*   dst = (const int*)d_in[2];

    const float* Wp[8]; const float* bp[8];
    for (int l = 0; l < 8; ++l) {
        Wp[l] = (const float*)d_in[3 + 2 * l];
        bp[l] = (const float*)d_in[4 + 2 * l];
    }
    // layer dims
    const int din_[8]  = {FF, HH, HH, HH, CC, HH, HH, HH};
    const int dout_[8] = {HH, HH, HH, CC, HH, HH, HH, FF};
    const int Kpad_[8] = {512, 256, 256, 256, 32, 256, 256, 256};
    const int Npad_[8] = {256, 256, 256, 16, 256, 256, 256, 512};

    // workspace layout (all sub-offsets 32B aligned)
    float* wsf       = (float*)d_ws;
    float* dinv      = wsf;                                  // N
    float* self_norm = wsf + NN;                             // N
    unsigned short* wswz = (unsigned short*)(wsf + 2 * NN);  // 8*131072 bf16
    const size_t WSLOT = 131072;
    float* buf0 = wsf + 2 * NN + (8 * WSLOT) / 2;            // N*500 (gemm out)
    float* buf1 = buf0 + (size_t)NN * FF;                    // N*500 (agg tmp)
    unsigned short* abuf_x = (unsigned short*)(buf1 + (size_t)NN * FF); // N*512 bf16
    unsigned short* abuf_h = abuf_x + (size_t)NN * 512;      // N*256 bf16
    unsigned short* abuf_z = abuf_h + (size_t)NN * HH;       // N*32  bf16

    // d_out layout: x_de | enc_h1 | enc_h2 | enc_h3 | z_en
    float* out_xde = (float*)d_out;
    float* out_e1  = out_xde + (size_t)NN * FF;
    float* out_e2  = out_e1  + (size_t)NN * HH;
    float* out_e3  = out_e2  + (size_t)NN * HH;
    float* out_z   = out_e3  + (size_t)NN * HH;

    const int TB = 256;

    // --- degree / normalization ---
    fill_ones_kernel<<<(NN + TB - 1) / TB, TB, 0, stream>>>(dinv, NN);
    deg_acc_kernel<<<(EE + TB - 1) / TB, TB, 0, stream>>>(dinv, dst, EE);
    deg_fin_kernel<<<(NN + TB - 1) / TB, TB, 0, stream>>>(dinv, self_norm, NN);

    // --- weight swizzle (fp32 -> bf16 fragment layout) ---
    for (int l = 0; l < 8; ++l) {
        int total = (Kpad_[l] >> 5) * (Npad_[l] >> 4) * 512;
        swizzle_w_kernel<<<(total + TB - 1) / TB, TB, 0, stream>>>(
            Wp[l], wswz + l * WSLOT, din_[l], dout_[l], Kpad_[l], Npad_[l]);
    }

    // --- pad/convert input activations ---
    pad_cvt_kernel<<<dim3(2, NN), TB, 0, stream>>>(abuf_x, x, FF, 512);

    auto run_layer = [&](int l, const unsigned short* inbf, int ldk,
                         float* aggbuf, float* relu32 /*null if none*/,
                         unsigned short* relubf /*null if none*/) {
        const int dout = dout_[l];
        const int Npad = Npad_[l];
        const int ktiles = Kpad_[l] >> 5;
        const int ntils = Npad >> 4;
        const int waves = (ntils + 1) / 2;
        dim3 g(NN / 16, (waves + 7) / 8);
        gemm_wmma_kernel<<<g, TB, 0, stream>>>(inbf, wswz + l * WSLOT, buf0,
                                               ldk, dout, ktiles, Npad);
        agg_init_kernel<<<dim3((dout + TB - 1) / TB, NN), TB, 0, stream>>>(
            aggbuf, buf0, bp[l], self_norm, dout);
        scatter_kernel<<<(unsigned)(((long long)EE * 32 + TB - 1) / TB), TB, 0, stream>>>(
            aggbuf, buf0, src, dst, dinv, EE, dout);
        if (relubf) {
            long long nd = (long long)NN * dout;   // dout == ldk_next == 256
            relu_cvt_kernel<<<(unsigned)((nd + TB - 1) / TB), TB, 0, stream>>>(
                relubf, relu32, relu32 ? 1 : 0, aggbuf, nd);
        }
    };

    run_layer(0, abuf_x, 512, buf1,    out_e1,  abuf_h); // enc1 -> enc_h1
    run_layer(1, abuf_h, 256, buf1,    out_e2,  abuf_h); // enc2 -> enc_h2
    run_layer(2, abuf_h, 256, buf1,    out_e3,  abuf_h); // enc3 -> enc_h3
    run_layer(3, abuf_h, 256, out_z,   nullptr, nullptr);// zen  -> z_en
    pad_cvt_kernel<<<dim3(1, NN), TB, 0, stream>>>(abuf_z, out_z, CC, 32);
    run_layer(4, abuf_z, 32,  buf1,    nullptr, abuf_h); // dec1
    run_layer(5, abuf_h, 256, buf1,    nullptr, abuf_h); // dec2
    run_layer(6, abuf_h, 256, buf1,    nullptr, abuf_h); // dec3
    run_layer(7, abuf_h, 256, out_xde, nullptr, nullptr);// xde  -> x_de
}
